// ConditionedSequenceFrameTransformerUpdate_2430951489767
// MI455X (gfx1250) — compile-verified, matrix operand order checked by reading, not run on hardware
//
#include <hip/hip_runtime.h>
#include <math.h>

#define DEV __device__ __forceinline__

typedef __attribute__((ext_vector_type(16))) _Float16 v16h;
typedef __attribute__((ext_vector_type(8)))  _Float16 v8h;
typedef __attribute__((ext_vector_type(8)))  float    v8f;

static constexpr int Nseq = 2048;
static constexpr int Cdim = 256;
static constexpr int H    = 8;
static constexpr int CH   = 32;
static constexpr int PQ   = 8;
static constexpr int PV   = 12;
static constexpr int BQ   = 32;
static constexpr int BK   = 128;
static constexpr int NBq  = 64;
static constexpr int PADW = 48;      // (BK-BQ)/2
static constexpr int PROJ_LD = 1440; // q 256 | k 256 | v 256 | qp 192 | kvp 480
static constexpr int FEAT = 704;     // o 256 | o_pt 288 | norms 96 | pair 64

// ---------------------------------------------------------------------------
// WMMA helpers (16x16x32 f16 -> f32), CDNA5 wave32 layouts (ISA 7.12.2)
// A-frag (and Bt-frag): lane l<16 -> row base+l, K {0..7,16..23};
//                       lane l+16 -> same row,  K {8..15,24..31}
// C/D: vgpr r -> M = r (lanes 0-15) or 8+r (lanes 16-31); N = lane&15
// ---------------------------------------------------------------------------
DEV v8f wmma(v16h a, v16h b, v8f c) {
  return __builtin_amdgcn_wmma_f32_16x16x32_f16(false, a, false, b, (short)0, c,
                                                false, false);
}

DEV v16h load_frag_h(const _Float16* base, int ld, int row0, int rowmax, int k0) {
  int lane = threadIdx.x & 31;
  int r = row0 + (lane & 15);
  if (r > rowmax) r = rowmax;
  const _Float16* p = base + (size_t)r * ld + k0 + ((lane >> 4) << 3);
  v16h f;
  ((v8h*)&f)[0] = *(const v8h*)(p);
  ((v8h*)&f)[1] = *(const v8h*)(p + 16);
  return f;
}

// per-lane row pointer variant (for gathered key rows)
DEV v16h load_frag_hp(const _Float16* rowbase) {
  const _Float16* p = rowbase + (((threadIdx.x & 31) >> 4) << 3);
  v16h f;
  ((v8h*)&f)[0] = *(const v8h*)(p);
  ((v8h*)&f)[1] = *(const v8h*)(p + 16);
  return f;
}

DEV void store_tile_f(float* Cp, int ldc, int m0, int n0, int M, int N, v8f acc) {
  int lane = threadIdx.x & 31;
  int n = n0 + (lane & 15);
  int mb = m0 + ((lane >> 4) << 3);
  if (n >= N) return;
#pragma unroll
  for (int r = 0; r < 8; ++r) { int m = mb + r; if (m < M) Cp[(size_t)m * ldc + n] = acc[r]; }
}
DEV void store_tile_h(_Float16* Cp, int ldc, int m0, int n0, int M, int N, v8f acc) {
  int lane = threadIdx.x & 31;
  int n = n0 + (lane & 15);
  int mb = m0 + ((lane >> 4) << 3);
  if (n >= N) return;
#pragma unroll
  for (int r = 0; r < 8; ++r) { int m = mb + r; if (m < M) Cp[(size_t)m * ldc + n] = (_Float16)acc[r]; }
}

// ---------------------------------------------------------------------------
// Generic WMMA GEMM: C[M,N](f32) = A[M,K](f16) * Bt[N,K](f16)^T
// Optional second f16 output image (Ch) for downstream WMMA consumers.
// block = 256 threads = 8 waves (2x4), block tile 64x128, wave tile 32x32
// ---------------------------------------------------------------------------
__global__ __launch_bounds__(256) void k_gemm(const _Float16* __restrict__ A,
                                              const _Float16* __restrict__ Bt,
                                              float* __restrict__ Cp,
                                              _Float16* __restrict__ Ch,
                                              int M, int N, int K,
                                              int lda, int ldb, int ldc) {
  int wave = threadIdx.x >> 5;
  int wm = wave >> 2, wn = wave & 3;
  int m0 = blockIdx.y * 64 + wm * 32;
  int n0 = blockIdx.x * 128 + wn * 32;
  if (m0 >= M || n0 >= N) return;  // wave-uniform
  v8f acc[2][2] = {};
  for (int k0 = 0; k0 < K; k0 += 32) {
    v16h a0 = load_frag_h(A, lda, m0,      M - 1, k0);
    v16h a1 = load_frag_h(A, lda, m0 + 16, M - 1, k0);
    v16h b0 = load_frag_h(Bt, ldb, n0,      N - 1, k0);
    v16h b1 = load_frag_h(Bt, ldb, n0 + 16, N - 1, k0);
    acc[0][0] = wmma(a0, b0, acc[0][0]);
    acc[0][1] = wmma(a0, b1, acc[0][1]);
    acc[1][0] = wmma(a1, b0, acc[1][0]);
    acc[1][1] = wmma(a1, b1, acc[1][1]);
  }
#pragma unroll
  for (int i = 0; i < 2; ++i)
#pragma unroll
    for (int j = 0; j < 2; ++j) {
      store_tile_f(Cp, ldc, m0 + 16 * i, n0 + 16 * j, M, N, acc[i][j]);
      if (Ch) store_tile_h(Ch, ldc, m0 + 16 * i, n0 + 16 * j, M, N, acc[i][j]);
    }
}

// ---------------------------------------------------------------------------
// Conversions
// ---------------------------------------------------------------------------
__global__ void k_cvt(const float* __restrict__ x, _Float16* __restrict__ y, int n) {
  int i = blockIdx.x * blockDim.x + threadIdx.x;
  if (i < n) y[i] = (_Float16)x[i];
}
// W[in,out] row-major f32  ->  Wt[out,in] row-major f16 (transpose + convert)
__global__ void k_cvtT(const float* __restrict__ W, _Float16* __restrict__ Wt,
                       int in_dim, int out_dim) {
  int idx = blockIdx.x * blockDim.x + threadIdx.x;
  if (idx >= in_dim * out_dim) return;
  int o = idx / in_dim, i = idx - o * in_dim;
  Wt[(size_t)o * in_dim + i] = (_Float16)W[(size_t)i * out_dim + o];
}

// ---------------------------------------------------------------------------
// s_mod = LN(s) * (cond@Wg + bg) + (cond@Wbeta + bbeta)    (gmbeta = cond@[Wg|Wbeta])
// ---------------------------------------------------------------------------
__global__ void k_mod(const float* __restrict__ s, const float* __restrict__ gmbeta,
                      const float* __restrict__ bg, const float* __restrict__ bbeta,
                      _Float16* __restrict__ smod) {
  int n = blockIdx.x * blockDim.x + threadIdx.x;
  if (n >= Nseq) return;
  const float* r = s + (size_t)n * Cdim;
  float mu = 0.f, s2 = 0.f;
  for (int c = 0; c < Cdim; ++c) { float v = r[c]; mu += v; s2 += v * v; }
  mu *= (1.f / Cdim);
  float inv = rsqrtf(s2 * (1.f / Cdim) - mu * mu + 1e-5f);
  const float* gb = gmbeta + (size_t)n * 512;
  for (int c = 0; c < Cdim; ++c) {
    float ln = (r[c] - mu) * inv;
    smod[(size_t)n * Cdim + c] =
        (_Float16)(ln * (gb[c] + bg[c]) + gb[256 + c] + bbeta[c]);
  }
}

// z LayerNorm over CZ=32 -> f16
__global__ void k_zln(const float* __restrict__ z, const float* __restrict__ gz,
                      const float* __restrict__ bz_, _Float16* __restrict__ zl,
                      int rows) {
  int rI = blockIdx.x * blockDim.x + threadIdx.x;
  if (rI >= rows) return;
  const float* r = z + (size_t)rI * 32;
  float mu = 0.f, s2 = 0.f;
#pragma unroll
  for (int c = 0; c < 32; ++c) { float v = r[c]; mu += v; s2 += v * v; }
  mu *= (1.f / 32.f);
  float inv = rsqrtf(s2 * (1.f / 32.f) - mu * mu + 1e-5f);
#pragma unroll
  for (int c = 0; c < 32; ++c)
    zl[(size_t)rI * 32 + c] = (_Float16)(((r[c] - mu) * inv) * gz[c] + bz_[c]);
}

// ---------------------------------------------------------------------------
// Per-row frame application: rotate q-points / kv-points, compute row norms.
// qpts/kpts: [n][h][32] f16 (24 coords + zero pad, ready as WMMA K=32 frags)
// vpts:      [n][h][36] f16 ; qn/kn: [n][h] f32
// ---------------------------------------------------------------------------
__global__ void k_rot(const float* __restrict__ proj, const float* __restrict__ rots,
                      const float* __restrict__ trans, _Float16* __restrict__ qpts,
                      _Float16* __restrict__ kpts, _Float16* __restrict__ vpts,
                      float* __restrict__ qn, float* __restrict__ kn) {
  int idx = blockIdx.x * blockDim.x + threadIdx.x;
  if (idx >= Nseq * H) return;
  int n = idx >> 3, h = idx & 7;
  float R[9];
#pragma unroll
  for (int i = 0; i < 9; ++i) R[i] = rots[(size_t)n * 9 + i];
  float t0 = trans[n * 3 + 0], t1 = trans[n * 3 + 1], t2 = trans[n * 3 + 2];

  const float* pq = proj + (size_t)n * PROJ_LD + 768 + h * (PQ * 3);
  _Float16* qo = qpts + ((size_t)n * H + h) * 32;
  float qs = 0.f;
#pragma unroll
  for (int p = 0; p < PQ; ++p) {
    float x = pq[p * 3], y = pq[p * 3 + 1], zz = pq[p * 3 + 2];
    float gx = R[0] * x + R[1] * y + R[2] * zz + t0;
    float gy = R[3] * x + R[4] * y + R[5] * zz + t1;
    float gz = R[6] * x + R[7] * y + R[8] * zz + t2;
    qo[p * 3] = (_Float16)gx; qo[p * 3 + 1] = (_Float16)gy; qo[p * 3 + 2] = (_Float16)gz;
    qs += gx * gx + gy * gy + gz * gz;
  }
#pragma unroll
  for (int c = PQ * 3; c < 32; ++c) qo[c] = (_Float16)0.f;
  qn[idx] = qs;

  const float* pk = proj + (size_t)n * PROJ_LD + 960 + h * ((PQ + PV) * 3);
  _Float16* ko = kpts + ((size_t)n * H + h) * 32;
  _Float16* vo = vpts + ((size_t)n * H + h) * (PV * 3);
  float ks = 0.f;
  for (int p = 0; p < PQ + PV; ++p) {
    float x = pk[p * 3], y = pk[p * 3 + 1], zz = pk[p * 3 + 2];
    float gx = R[0] * x + R[1] * y + R[2] * zz + t0;
    float gy = R[3] * x + R[4] * y + R[5] * zz + t1;
    float gz = R[6] * x + R[7] * y + R[8] * zz + t2;
    if (p < PQ) {
      ko[p * 3] = (_Float16)gx; ko[p * 3 + 1] = (_Float16)gy; ko[p * 3 + 2] = (_Float16)gz;
      ks += gx * gx + gy * gy + gz * gz;
    } else {
      _Float16* d = vo + (p - PQ) * 3;
      d[0] = (_Float16)gx; d[1] = (_Float16)gy; d[2] = (_Float16)gz;
    }
  }
#pragma unroll
  for (int c = PQ * 3; c < 32; ++c) ko[c] = (_Float16)0.f;
  kn[idx] = ks;
}

// ---------------------------------------------------------------------------
// Fused IPA attention: one workgroup per (block nb, head h); 2 waves,
// each wave owns 16 query rows.  Scores = QK^T/sqrt(96) + sqrt(1/3) b
//   - hw/2 (|q|^2 + |k|^2 - 2 q.k_pts)  + mask ; softmax ; 2 AV WMMA GEMMs + VALU pair sum.
// All Q/K/V operands are f16 images; fragment loads are pure b128.
// ---------------------------------------------------------------------------
__global__ __launch_bounds__(64) void k_attn(
    const _Float16* __restrict__ proj_h, const _Float16* __restrict__ qpts,
    const _Float16* __restrict__ kpts, const _Float16* __restrict__ vpts,
    const float* __restrict__ qn, const float* __restrict__ kn,
    const float* __restrict__ bz, const float* __restrict__ s_mask,
    const float* __restrict__ head_weights, const float* __restrict__ rots,
    const float* __restrict__ trans, _Float16* __restrict__ feat) {
  int nb = blockIdx.x >> 3;
  int h  = blockIdx.x & 7;
  int wave = threadIdx.x >> 5;
  int lane = threadIdx.x & 31;

  __shared__ _Float16 vT[32][128];    // V^T   (c , gathered k)
  __shared__ _Float16 vpT[48][128];   // Vpts^T (dim<36 , gathered k), zero-pad
  __shared__ float    sc[2][16 * 128];// per-wave scores (f32) -> probs (f16 in place)
  __shared__ float    opt[2][16][40]; // raw o_pt (global frame)

  const float hw = log1pf(expf(head_weights[h])) * 0.09622504486493763f; // sqrt(1/108)
  const float sc_qk = 0.10206207261596575f;  // 1/sqrt(96)
  const float sc_b  = 0.5773502691896258f;   // sqrt(1/3)

  // ---- stage transposed V / Vpts tiles (gathered, invalid -> 0) ----
  for (int i = threadIdx.x; i < 32 * 128; i += 64) {
    int c = i >> 7, kk = i & 127;
    int src = nb * BQ + kk - PADW;
    _Float16 v = (_Float16)0.f;
    if (src >= 0 && src < Nseq) v = proj_h[(size_t)src * PROJ_LD + 512 + h * CH + c];
    vT[c][kk] = v;
  }
  for (int i = threadIdx.x; i < 48 * 128; i += 64) {
    int d = i >> 7, kk = i & 127;
    int src = nb * BQ + kk - PADW;
    _Float16 v = (_Float16)0.f;
    if (d < 36 && src >= 0 && src < Nseq) v = vpts[((size_t)src * H + h) * 36 + d];
    vpT[d][kk] = v;
  }
  __syncthreads();

  const int m0 = wave * 16;                 // this wave's query rows in block
  float* S = sc[wave];

  // ---- scores: 8 N-tiles of 16x16, each = 2 WMMAs + epilogue ----
  for (int nt = 0; nt < 8; ++nt) {
    int n0 = nt * 16;
    v16h aQ = load_frag_h(proj_h + h * CH, PROJ_LD, nb * BQ + m0, Nseq - 1, 0);
    v16h aP = load_frag_h(qpts + h * 32, H * 32, nb * BQ + m0, Nseq - 1, 0);

    int kk = n0 + (lane & 15);
    int srcL = nb * BQ + kk - PADW;
    int srcC = srcL < 0 ? 0 : (srcL > Nseq - 1 ? Nseq - 1 : srcL);
    v16h bK = load_frag_hp(proj_h + (size_t)srcC * PROJ_LD + 256 + h * CH);
    v16h bP = load_frag_hp(kpts + ((size_t)srcC * H + h) * 32);

    v8f zc = {};
    v8f aQK = wmma(aQ, bK, zc);
    v8f aPT = wmma(aP, bP, zc);

    bool validk = (srcL >= 0) && (srcL < Nseq);
    float mk  = validk ? s_mask[srcL] : 0.f;
    float knv = validk ? kn[(size_t)srcL * H + h] : 0.f;
    int mbase = (lane >> 4) << 3;
#pragma unroll
    for (int r = 0; r < 8; ++r) {
      int m = mbase + r;              // 0..15 within wave tile
      int nq = nb * BQ + m0 + m;
      float bval = bz[((size_t)nq * BK + kk) * 16 + h];
      float qnv  = qn[(size_t)nq * H + h];
      float sv = aQK[r] * sc_qk + sc_b * bval
               - 0.5f * hw * (qnv + knv - 2.f * aPT[r])
               + 1e8f * (s_mask[nq] * mk - 1.f);
      S[m * 128 + kk] = sv;
    }
  }

  // ---- softmax per row (wave-local), write probs as f16 in place ----
  _Float16* P = (_Float16*)S;
  for (int q = 0; q < 16; ++q) {
    float v0 = S[q * 128 + lane], v1 = S[q * 128 + lane + 32];
    float v2 = S[q * 128 + lane + 64], v3 = S[q * 128 + lane + 96];
    float mx = fmaxf(fmaxf(v0, v1), fmaxf(v2, v3));
    for (int off = 16; off > 0; off >>= 1) mx = fmaxf(mx, __shfl_xor(mx, off, 32));
    float e0 = expf(v0 - mx), e1 = expf(v1 - mx);
    float e2 = expf(v2 - mx), e3 = expf(v3 - mx);
    float sm = e0 + e1 + e2 + e3;
    for (int off = 16; off > 0; off >>= 1) sm += __shfl_xor(sm, off, 32);
    float inv = 1.f / sm;
    P[q * 128 + lane]      = (_Float16)(e0 * inv);
    P[q * 128 + lane + 32] = (_Float16)(e1 * inv);
    P[q * 128 + lane + 64] = (_Float16)(e2 * inv);
    P[q * 128 + lane + 96] = (_Float16)(e3 * inv);
  }

  // ---- o = probs @ V  (16x32 per wave, K=128) ----
  for (int nt = 0; nt < 2; ++nt) {
    v8f acc = {};
    for (int k0 = 0; k0 < 128; k0 += 32) {
      v16h a = load_frag_h(P, 128, 0, 15, k0);
      v16h b = load_frag_h(&vT[0][0], 128, nt * 16, 31, k0);
      acc = wmma(a, b, acc);
    }
    store_tile_h(feat + h * CH, FEAT, nb * BQ + m0, nt * 16, Nseq, CH, acc);
  }

  // ---- o_pt (global frame) = probs @ Vpts  (16x48, only 36 used) ----
  for (int nt = 0; nt < 3; ++nt) {
    v8f acc = {};
    for (int k0 = 0; k0 < 128; k0 += 32) {
      v16h a = load_frag_h(P, 128, 0, 15, k0);
      v16h b = load_frag_h(&vpT[0][0], 128, nt * 16, 47, k0);
      acc = wmma(a, b, acc);
    }
    int n = nt * 16 + (lane & 15);
    int mb = (lane >> 4) << 3;
    if (n < 40) {
#pragma unroll
      for (int r = 0; r < 8; ++r) opt[wave][mb + r][n] = acc[r];
    }
  }

  // ---- back-rotate o_pt into local frame + norms ----
  for (int tsk = lane; tsk < 16 * PV; tsk += 32) {
    int q = tsk / PV, pv = tsk - (tsk / PV) * PV;
    int nq = nb * BQ + m0 + q;
    float R[9];
#pragma unroll
    for (int i = 0; i < 9; ++i) R[i] = rots[(size_t)nq * 9 + i];
    float dx = opt[wave][q][pv * 3 + 0] - trans[nq * 3 + 0];
    float dy = opt[wave][q][pv * 3 + 1] - trans[nq * 3 + 1];
    float dz = opt[wave][q][pv * 3 + 2] - trans[nq * 3 + 2];
    float lx = R[0] * dx + R[3] * dy + R[6] * dz;   // R^T d
    float ly = R[1] * dx + R[4] * dy + R[7] * dz;
    float lz = R[2] * dx + R[5] * dy + R[8] * dz;
    _Float16* F = feat + (size_t)nq * FEAT;
    F[256 + (h * PV + pv) * 3 + 0] = (_Float16)lx;
    F[256 + (h * PV + pv) * 3 + 1] = (_Float16)ly;
    F[256 + (h * PV + pv) * 3 + 2] = (_Float16)lz;
    F[544 + h * PV + pv] = (_Float16)sqrtf(lx * lx + ly * ly + lz * lz + 1e-8f);
  }

  // ---- o_pair[q,c] = sum_k probs[q,k] * pair_z[q,k,c]  (c<8, VALU) ----
  for (int q = 0; q < 16; ++q) {
    int nq = nb * BQ + m0 + q;
    float acc[8] = {};
    for (int k = lane; k < 128; k += 32) {
      float p = (float)P[q * 128 + k];
      const float* pz = bz + ((size_t)nq * BK + k) * 16 + 8;
#pragma unroll
      for (int c = 0; c < 8; ++c) acc[c] += p * pz[c];
    }
#pragma unroll
    for (int c = 0; c < 8; ++c) {
      float v = acc[c];
      for (int off = 16; off > 0; off >>= 1) v += __shfl_xor(v, off, 32);
      if (lane == 0) feat[(size_t)nq * FEAT + 640 + h * 8 + c] = (_Float16)v;
    }
  }
}

// ---------------------------------------------------------------------------
extern "C" void kernel_launch(void* const* d_in, const int* in_sizes, int n_in,
                              void* d_out, int out_size, void* d_ws, size_t ws_size,
                              hipStream_t stream) {
  const float* s      = (const float*)d_in[0];
  const float* cond   = (const float*)d_in[1];
  const float* z      = (const float*)d_in[2];
  const float* trans  = (const float*)d_in[3];
  const float* rots   = (const float*)d_in[4];
  const float* s_mask = (const float*)d_in[5];
  const float* Wq     = (const float*)d_in[6];
  const float* Wk     = (const float*)d_in[7];
  const float* Wv     = (const float*)d_in[8];
  const float* Wqp    = (const float*)d_in[9];
  const float* Wkvp   = (const float*)d_in[10];
  const float* Wb     = (const float*)d_in[11];
  const float* Wdz    = (const float*)d_in[12];
  const float* hwghts = (const float*)d_in[13];
  const float* Wout   = (const float*)d_in[14];
  const float* Wg     = (const float*)d_in[15];
  const float* bg     = (const float*)d_in[16];
  const float* Wbeta  = (const float*)d_in[17];
  const float* bbeta  = (const float*)d_in[18];
  const float* g_z    = (const float*)d_in[19];
  const float* b_z    = (const float*)d_in[20];
  float* out = (float*)d_out;

  const int ZROWS = NBq * BQ * BK;  // 262144

  char* base = (char*)d_ws;
  size_t off = 0;
  auto alloc = [&](size_t bytes) -> void* {
    void* p = base + off;
    off = (off + bytes + 255) & ~(size_t)255;
    return p;
  };
  _Float16* cond_h  = (_Float16*)alloc((size_t)Nseq * Cdim * 2);
  _Float16* smod_h  = (_Float16*)alloc((size_t)Nseq * Cdim * 2);
  _Float16* Wgb_t   = (_Float16*)alloc((size_t)512 * 256 * 2);
  _Float16* Wproj_t = (_Float16*)alloc((size_t)PROJ_LD * 256 * 2);
  _Float16* Wbz_t   = (_Float16*)alloc((size_t)32 * 32 * 2);
  _Float16* Wout_t  = (_Float16*)alloc((size_t)256 * FEAT * 2);
  float*    gmbeta  = (float*)   alloc((size_t)Nseq * 512 * 4);
  float*    proj    = (float*)   alloc((size_t)Nseq * PROJ_LD * 4);
  _Float16* proj_h  = (_Float16*)alloc((size_t)Nseq * PROJ_LD * 2);
  _Float16* z_ln    = (_Float16*)alloc((size_t)ZROWS * 32 * 2);
  float*    bz      = (float*)   alloc((size_t)ZROWS * 16 * 4);
  _Float16* qpts    = (_Float16*)alloc((size_t)Nseq * H * 32 * 2);
  _Float16* kpts    = (_Float16*)alloc((size_t)Nseq * H * 32 * 2);
  _Float16* vpts    = (_Float16*)alloc((size_t)Nseq * H * 36 * 2);
  float*    qn      = (float*)   alloc((size_t)Nseq * H * 4);
  float*    kn      = (float*)   alloc((size_t)Nseq * H * 4);
  _Float16* feat    = (_Float16*)alloc((size_t)Nseq * FEAT * 2);

  auto gemm = [&](const _Float16* A, const _Float16* Bt, float* Cp, _Float16* Ch,
                  int M, int N, int K, int lda, int ldb, int ldc) {
    dim3 grid((N + 127) / 128, (M + 63) / 64);
    k_gemm<<<grid, 256, 0, stream>>>(A, Bt, Cp, Ch, M, N, K, lda, ldb, ldc);
  };
  auto cvtT = [&](const float* W, _Float16* Wt, int in_dim, int out_dim) {
    int n = in_dim * out_dim;
    k_cvtT<<<(n + 255) / 256, 256, 0, stream>>>(W, Wt, in_dim, out_dim);
  };

  // weight prep
  hipMemsetAsync(Wbz_t, 0, (size_t)32 * 32 * 2, stream);
  k_cvt<<<(Nseq * Cdim + 255) / 256, 256, 0, stream>>>(cond, cond_h, Nseq * Cdim);
  cvtT(Wg,    Wgb_t,                 256, 256);
  cvtT(Wbeta, Wgb_t + 256 * 256,     256, 256);
  cvtT(Wq,    Wproj_t,               256, 256);
  cvtT(Wk,    Wproj_t + 256 * 256,   256, 256);
  cvtT(Wv,    Wproj_t + 512 * 256,   256, 256);
  cvtT(Wqp,   Wproj_t + 768 * 256,   256, 192);
  cvtT(Wkvp,  Wproj_t + 960 * 256,   256, 480);
  cvtT(Wb,    Wbz_t,                  32,   8);
  cvtT(Wdz,   Wbz_t + 8 * 32,         32,   8);
  cvtT(Wout,  Wout_t,                FEAT, 256);

  // modulation: gmbeta = cond @ [Wg|Wbeta] ; s_mod = LN(s)*g + beta
  gemm(cond_h, Wgb_t, gmbeta, nullptr, Nseq, 512, 256, 256, 256, 512);
  k_mod<<<(Nseq + 255) / 256, 256, 0, stream>>>(s, gmbeta, bg, bbeta, smod_h);

  // projections (q|k|v|qp|kvp in one GEMM), dual f32 + f16 outputs
  gemm(smod_h, Wproj_t, proj, proj_h, Nseq, PROJ_LD, 256, 256, 256, PROJ_LD);

  // z path: LN -> [b | pair_z] GEMM
  k_zln<<<(ZROWS + 255) / 256, 256, 0, stream>>>(z, g_z, b_z, z_ln, ZROWS);
  gemm(z_ln, Wbz_t, bz, nullptr, ZROWS, 16, 32, 32, 32, 16);

  // frames
  k_rot<<<(Nseq * H + 255) / 256, 256, 0, stream>>>(proj, rots, trans, qpts, kpts,
                                                    vpts, qn, kn);

  // fused attention, one block per (nb, head)
  k_attn<<<NBq * H, 64, 0, stream>>>(proj_h, qpts, kpts, vpts, qn, kn, bz, s_mask,
                                     hwghts, rots, trans, feat);

  // output projection
  gemm(feat, Wout_t, out, nullptr, Nseq, 256, FEAT, FEAT, FEAT, 256);

  (void)in_sizes; (void)n_in; (void)out_size; (void)ws_size;
}
